// GRULayer_6700148982668
// MI455X (gfx1250) — compile-verified, matrix operand order checked by reading, not run on hardware
//
#include <hip/hip_runtime.h>

#define T_STEPS 2048
#define BATCH   64
#define IN      256
#define HID     256
#define G3      768   // 3*HID
#define WELEM   (G3 * HID)           // 196608 elements per weight matrix

typedef __attribute__((ext_vector_type(16))) __bf16 bf16x16;
typedef __attribute__((ext_vector_type(8)))  float  f32x8;
typedef __attribute__((ext_vector_type(4)))  unsigned u32x4;
typedef __attribute__((ext_vector_type(8)))  int      i32x8;
typedef __attribute__((ext_vector_type(4)))  int      i32x4;

#if defined(__has_builtin)
#if __has_builtin(__builtin_amdgcn_tensor_load_to_lds) && \
    __has_builtin(__builtin_amdgcn_s_wait_tensorcnt)
#define HAVE_TDM 1
#endif
#endif

// Exact round-to-nearest-even f32->bf16 (cold paths).
static __device__ __forceinline__ unsigned bfbits(float f) {
  union { float f; unsigned u; } c; c.f = f;
  unsigned u = c.u;
  return (u + 0x7FFFu + ((u >> 16) & 1u)) >> 16;
}

// Fast pack of two f32 into a packed bf16 pair: v_perm_b32 + two adds.
static __device__ __forceinline__ unsigned pk2f(float lo, float hi) {
  union { float f; unsigned u; } a, b; a.f = lo; b.f = hi;
  return __builtin_amdgcn_perm(b.u + 0x8000u, a.u + 0x8000u, 0x07060302u);
}

union Frag { bf16x16 v; unsigned u[8]; uint4 q[2]; };

// 16x32 bf16 WMMA A/B fragment from a row-major fp32 matrix (ld in elements).
// ISA layout: lane r (0..15) holds K {0..7,16..23}; lane r+16 holds K {8..15,24..31}.
static __device__ __forceinline__ bf16x16
load_frag_f32(const float* __restrict__ base, int ld, int r, int hi) {
  const float* p = base + r * ld + hi * 8;
  float4 a = *(const float4*)(p);
  float4 b = *(const float4*)(p + 4);
  float4 c = *(const float4*)(p + 16);
  float4 d = *(const float4*)(p + 20);
  Frag f;
  f.u[0] = pk2f(a.x, a.y); f.u[1] = pk2f(a.z, a.w);
  f.u[2] = pk2f(b.x, b.y); f.u[3] = pk2f(b.z, b.w);
  f.u[4] = pk2f(c.x, c.y); f.u[5] = pk2f(c.z, c.w);
  f.u[6] = pk2f(d.x, d.y); f.u[7] = pk2f(d.z, d.w);
  return f.v;
}

// Same fragment straight from a row-major bf16 matrix (global or LDS).
static __device__ __forceinline__ bf16x16
load_frag_bf16(const unsigned short* base, int ld, int k0, int r, int hi) {
  const unsigned short* p = base + r * ld + k0 + hi * 8;
  Frag f;
  f.q[0] = *(const uint4*)(p);
  f.q[1] = *(const uint4*)(p + 16);
  return f.v;
}

// Tensor Data Mover: DMA a 2D tile [rows=16 x cols=768 fp32] global -> LDS.
// D# per CDNA5 ISA ch.8: group0 = count/lds_addr/global_addr/type,
// group1 = data_size + tensor dims + tile dims + strides. Groups 2/3 = 0 (2D).
static __device__ __forceinline__ void tdm_stage_gi(const float* src, unsigned ldsOff) {
#if defined(HAVE_TDM)
  unsigned long long ga = (unsigned long long)(uintptr_t)src;
  u32x4 g0;
  g0[0] = 1u;                                              // count=1, user descriptor
  g0[1] = ldsOff;                                          // lds_addr (bytes)
  g0[2] = (unsigned)ga;                                    // global_addr[31:0]
  g0[3] = (unsigned)((ga >> 32) & 0x01FFFFFFull) | (2u << 30);  // addr[56:32] | type=2
  i32x8 g1;
  g1[0] = (int)(2u << 16);        // data_size=2 (4 bytes); no multicast/pad/iterate
  g1[1] = (int)(768u << 16);      // tensor_dim0[15:0] at bits 63:48
  g1[2] = (int)(16u << 16);       // tensor_dim0 hi = 0 ; tensor_dim1[15:0] = 16
  g1[3] = (int)(768u << 16);      // tensor_dim1 hi = 0 ; tile_dim0 = 768
  g1[4] = 16;                     // tile_dim1 = 16 ; tile_dim2 = 0
  g1[5] = 768;                    // tensor_dim0_stride[31:0] = 768 elements
  g1[6] = 0;                      // stride hi / dim1_stride lo
  g1[7] = 0;
  i32x4 z4 = {0, 0, 0, 0};
#if __clang_major__ >= 23
  i32x8 z8 = {0, 0, 0, 0, 0, 0, 0, 0};
  __builtin_amdgcn_tensor_load_to_lds(g0, g1, z4, z4, z8, 0);
#else
  __builtin_amdgcn_tensor_load_to_lds(g0, g1, z4, z4, 0);
#endif
#else
  (void)src; (void)ldsOff;
#endif
}

// ---------------------------------------------------------------------------
// Kernel 0: one-time RNE conversion of W_ih and W_hh to bf16 (packed pairs).
// ---------------------------------------------------------------------------
__global__ __launch_bounds__(256) void cvt_w_kernel(
    const float* __restrict__ wih, const float* __restrict__ whh,
    unsigned* __restrict__ dst) {
  const int i = blockIdx.x * 256 + threadIdx.x;          // dword index
  const int half = WELEM / 2;                            // 98304
  const float* src = (i < half) ? wih : whh;
  const int k = (i < half) ? i : i - half;
  dst[i] = bfbits(src[2 * k]) | (bfbits(src[2 * k + 1]) << 16);
}

// ---------------------------------------------------------------------------
// Kernel 1: gi = x @ W_ih^T + b_ih  (M = tc*B, N = 768, K = 256)
// One WG (512 thr / 16 waves) per 64-row M-block; each wave: 3 N-tiles,
// B fragments held in registers across 4 M-tiles.
// ---------------------------------------------------------------------------
__global__ __launch_bounds__(512) void gi_gemm_kernel(
    const float* __restrict__ x, const unsigned short* __restrict__ WihB,
    const float* __restrict__ bih, float* __restrict__ gi) {
  const int tid = threadIdx.x, wave = tid >> 5, lane = tid & 31;
  const int r = lane & 15, hi = lane >> 4;
  const int row_base = blockIdx.x * 64;

#pragma unroll
  for (int nti = 0; nti < 3; ++nti) {
    const int nt = wave * 3 + nti;
    bf16x16 Bf[8];
#pragma unroll
    for (int kc = 0; kc < 8; ++kc)
      Bf[kc] = load_frag_bf16(WihB + (size_t)(nt * 16) * IN, IN, kc * 32, r, hi);
    const float bias = bih[nt * 16 + r];

#pragma unroll
    for (int mt = 0; mt < 4; ++mt) {
      const int row0 = row_base + mt * 16;
      bf16x16 A[8];
#pragma unroll
      for (int kc = 0; kc < 8; ++kc)
        A[kc] = load_frag_f32(x + (size_t)row0 * IN + kc * 32, IN, r, hi);
      f32x8 acc = {0.f, 0.f, 0.f, 0.f, 0.f, 0.f, 0.f, 0.f};
#pragma unroll
      for (int kc = 0; kc < 8; ++kc)
        acc = __builtin_amdgcn_wmma_f32_16x16x32_bf16(false, A[kc], false, Bf[kc],
                                                      (short)0, acc, false, false);
#pragma unroll
      for (int v = 0; v < 8; ++v)
        gi[(size_t)(row0 + hi * 8 + v) * G3 + nt * 16 + r] = acc[v] + bias;
    }
  }
}

// ---------------------------------------------------------------------------
// Kernel 2: sequential GRU scan over a chunk of tc steps.
// Grid = 4 WGs (16 batch rows each, independent recurrences).
// W_hh resident in VGPRs as bf16 fragments. Per step, the Tensor Data Mover
// DMAs the next gi tile [16 x 768] into LDS while WMMA runs; s_wait_tensorcnt
// + workgroup barrier publish it before the gate phase.
// ---------------------------------------------------------------------------
__global__ __launch_bounds__(512) void gru_scan_kernel(
    const float* __restrict__ gi, const unsigned short* __restrict__ WhhB,
    const float* __restrict__ bhh, float* __restrict__ out,
    float* __restrict__ hstate, int t0, int tc, int first) {
  __shared__ float          ghL[16][772];      // pad 772 -> conflict-free stores
  __shared__ float          giL[16][768];      // TDM staging for current step's gi
  __shared__ unsigned short hbfL[16][264];     // bf16 h in fragment-ready layout
  __shared__ float          bhhL[G3];

  const int tid = threadIdx.x, wave = tid >> 5, lane = tid & 31;
  const int r = lane & 15, hi = lane >> 4;
  const int b0 = blockIdx.x * 16;
  const unsigned giLoff = (unsigned)(uintptr_t)&giL[0][0];  // LDS byte offset

  // Persistently register-resident recurrent weights (bf16 fragments).
  bf16x16 Bw[3][8];
#pragma unroll
  for (int nti = 0; nti < 3; ++nti) {
    const int nt = wave * 3 + nti;
#pragma unroll
    for (int kc = 0; kc < 8; ++kc)
      Bw[nti][kc] = load_frag_bf16(WhhB + (size_t)(nt * 16) * HID, HID, kc * 32, r, hi);
  }

  for (int i = tid; i < G3; i += 512) bhhL[i] = bhh[i];
  if (first) {
    for (int i = tid; i < 16 * 264; i += 512) ((unsigned short*)hbfL)[i] = 0;
  } else {
    for (int i = tid; i < 16 * HID; i += 512) {
      int b = i >> 8, j = i & 255;
      hbfL[b][j] = (unsigned short)bfbits(hstate[(size_t)(b0 + b) * HID + j]);
    }
  }

  // Kick off the DMA for step 0 (lands under the first GEMM).
#if defined(HAVE_TDM)
  if (tid == 0) tdm_stage_gi(gi + (size_t)b0 * G3, giLoff);
#else
  for (int i = tid * 4; i < 16 * G3; i += 512 * 4)
    *(float4*)(&giL[0][0] + i) = *(const float4*)(gi + (size_t)b0 * G3 + i);
#endif
  __syncthreads();

  const int eb = tid >> 5;          // elementwise phase: wave == batch row
  const int j0 = lane * 8;          // 8 columns per lane

  for (int t = 0; t < tc; ++t) {
    // Warm L2 one step ahead of the DMA engine (fire-and-forget prefetch).
    if (t + 1 < tc) {
      const float* gq = gi + (size_t)((t + 1) * BATCH + b0 + eb) * G3;
      __builtin_prefetch(gq + j0, 0, 0);
      __builtin_prefetch(gq + 256 + j0, 0, 0);
      __builtin_prefetch(gq + 512 + j0, 0, 0);
    }

    // gh = h @ W_hh^T   (3 independent accumulation chains per wave)
    f32x8 acc[3];
#pragma unroll
    for (int nti = 0; nti < 3; ++nti)
#pragma unroll
      for (int v = 0; v < 8; ++v) acc[nti][v] = 0.f;
#pragma unroll
    for (int kc = 0; kc < 8; ++kc) {
      bf16x16 Af = load_frag_bf16(&hbfL[0][0], 264, kc * 32, r, hi);
#pragma unroll
      for (int nti = 0; nti < 3; ++nti)
        acc[nti] = __builtin_amdgcn_wmma_f32_16x16x32_bf16(false, Af, false, Bw[nti][kc],
                                                           (short)0, acc[nti], false, false);
    }
#pragma unroll
    for (int nti = 0; nti < 3; ++nti) {
      const int nc = (wave * 3 + nti) * 16 + r;
#pragma unroll
      for (int v = 0; v < 8; ++v) ghL[hi * 8 + v][nc] = acc[nti][v];
    }

    // Ensure this step's gi DMA has landed, then publish via the barrier.
#if defined(HAVE_TDM)
    __builtin_amdgcn_s_wait_tensorcnt(0);
#endif
    __syncthreads();

    // Gates + state update; each thread owns (row eb, cols j0..j0+7).
    float hnew[8];
#pragma unroll
    for (int u = 0; u < 8; ++u) {
      const int j = j0 + u;
      float ir = giL[eb][j];
      float iz = giL[eb][256 + j];
      float in_ = giL[eb][512 + j];
      float hr = ghL[eb][j]       + bhhL[j];
      float hz = ghL[eb][256 + j] + bhhL[256 + j];
      float hn = ghL[eb][512 + j] + bhhL[512 + j];
      float hprev = __uint_as_float(((unsigned)hbfL[eb][j]) << 16);
      float rg = 1.f / (1.f + __expf(-(ir + hr)));
      float zg = 1.f / (1.f + __expf(-(iz + hz)));
      float ng = tanhf(in_ + rg * hn);
      hnew[u] = (1.f - zg) * ng + zg * hprev;
    }
    const size_t orow = (size_t)((t0 + t) * BATCH + b0 + eb) * HID + j0;
    *(float4*)(out + orow)     = make_float4(hnew[0], hnew[1], hnew[2], hnew[3]);
    *(float4*)(out + orow + 4) = make_float4(hnew[4], hnew[5], hnew[6], hnew[7]);
#pragma unroll
    for (int u = 0; u < 8; u += 2)
      *(unsigned*)&hbfL[eb][j0 + u] = pk2f(hnew[u], hnew[u + 1]);
    __syncthreads();   // all giL readers done; safe to restage

    // Issue the DMA for step t+1; it overlaps the next WMMA burst.
    if (t + 1 < tc) {
#if defined(HAVE_TDM)
      if (tid == 0)
        tdm_stage_gi(gi + (size_t)((t + 1) * BATCH + b0) * G3, giLoff);
#else
      for (int i = tid * 4; i < 16 * G3; i += 512 * 4)
        *(float4*)(&giL[0][0] + i) =
            *(const float4*)(gi + (size_t)((t + 1) * BATCH + b0) * G3 + i);
      __syncthreads();
#endif
    }
  }

  // Persist running state (== h_last after the final chunk).
  for (int i = tid; i < 16 * HID; i += 512) {
    int b = i >> 8, j = i & 255;
    hstate[(size_t)(b0 + b) * HID + j] = __uint_as_float(((unsigned)hbfL[b][j]) << 16);
  }
}

extern "C" void kernel_launch(void* const* d_in, const int* in_sizes, int n_in,
                              void* d_out, int out_size, void* d_ws, size_t ws_size,
                              hipStream_t stream) {
  (void)in_sizes; (void)n_in; (void)out_size;
  const float* x   = (const float*)d_in[0];
  const float* Wih = (const float*)d_in[1];
  const float* Whh = (const float*)d_in[2];
  const float* bih = (const float*)d_in[3];
  const float* bhh = (const float*)d_in[4];
  float* out    = (float*)d_out;
  float* hstate = out + (size_t)T_STEPS * BATCH * HID;   // h_last slot = running state

  // Workspace layout: [W_ih bf16 | W_hh bf16 | gi chunk]
  unsigned short* WihB = (unsigned short*)d_ws;
  unsigned short* WhhB = WihB + WELEM;
  const size_t wBytes  = (size_t)2 * WELEM * sizeof(unsigned short);  // 768 KB
  float* gi = (float*)((char*)d_ws + wBytes);

  const size_t stepBytes = (size_t)BATCH * G3 * sizeof(float);        // 192 KB / step
  size_t giCap = (ws_size > wBytes) ? (ws_size - wBytes) : 0;
  int Tc = (int)(giCap / stepBytes);
  if (Tc < 1) Tc = 1;
  if (Tc > T_STEPS) Tc = T_STEPS;

  cvt_w_kernel<<<WELEM / 256, 256, 0, stream>>>(Wih, Whh, (unsigned*)d_ws);

  for (int t0 = 0; t0 < T_STEPS; t0 += Tc) {
    const int tc = (T_STEPS - t0 < Tc) ? (T_STEPS - t0) : Tc;
    gi_gemm_kernel<<<(tc * BATCH) / 64, 512, 0, stream>>>(
        x + (size_t)t0 * BATCH * IN, WihB, bih, gi);
    gru_scan_kernel<<<BATCH / 16, 512, 0, stream>>>(
        gi, WhhB, bhh, out, hstate, t0, tc, t0 == 0 ? 1 : 0);
  }
}
